// MultiHeadAttention_72937134621107
// MI455X (gfx1250) — matrix operand order repaired and checked by reading, not compiled, not run on hardware
//
#include <hip/hip_runtime.h>
#include <hip/hip_bf16.h>

// ---------------------------------------------------------------------------
// MI455X (gfx1250) multi-head attention forward, WMMA bf16 everywhere.
//   1. cvt:       W_Q/W_K/W_V/W_fc f32 -> bf16 (ws)
//   2. proj:      Q = x_q@Wq^T -> [b,h,s,d] bf16; K likewise; V -> [b,h,d,s]
//   3. fused attn: scores + softmax + P@V in ONE kernel.
//                  Q tile staged into LDS via TDM (tensor_load_to_lds);
//                  attn written to d_out exactly once (f32, required output);
//                  P staged in LDS (bf16) and consumed by context WMMAs.
//   4. outproj:   pre = ctx @ Wfc^T -> f32 (ws)
//   5. ln:        out = LayerNorm(pre + x_q) -> d_out
// ---------------------------------------------------------------------------

typedef __attribute__((ext_vector_type(16))) __bf16 v16bf;
typedef __attribute__((ext_vector_type(8)))  __bf16 v8bf;
typedef __attribute__((ext_vector_type(8)))  float  v8f;
typedef __attribute__((ext_vector_type(4)))  unsigned int v4u;
typedef __attribute__((ext_vector_type(8)))  int  v8i;
typedef __attribute__((ext_vector_type(4)))  int  v4i;

__device__ __forceinline__ v8f wmma_bf16(v16bf a, v16bf b, v8f c) {
  return __builtin_amdgcn_wmma_f32_16x16x32_bf16(false, a, false, b, (short)0, c,
                                                 false, false);
}

// ---- fragment loaders (CDNA5 ISA 7.12.2, wave32) ---------------------------
// A (16x32 bf16): lane holds row m=l&15; elems 0..7 -> K=kb..kb+7,
// elems 8..15 -> K=16+kb.., kb=(l>>4)*8.
__device__ __forceinline__ v16bf load_a_bf16(const __bf16* base,
                                             int row0, int ld, int k0, int lane) {
  int m  = lane & 15;
  int kb = (lane >> 4) * 8;
  const __bf16* p = base + (size_t)(row0 + m) * ld + k0;
  union { v16bf v; v8bf h[2]; } u;
  u.h[0] = *(const v8bf*)(p + kb);
  u.h[1] = *(const v8bf*)(p + 16 + kb);
  return u.v;
}

__device__ __forceinline__ v16bf load_a_f32(const float* __restrict__ base,
                                            int row0, int ld, int k0, int lane) {
  int m  = lane & 15;
  int kb = (lane >> 4) * 8;
  const float* p = base + (size_t)(row0 + m) * ld + k0;
  union { v16bf v; __bf16 e[16]; } u;
#pragma unroll
  for (int i = 0; i < 8; ++i) u.e[i]     = (__bf16)p[kb + i];
#pragma unroll
  for (int i = 0; i < 8; ++i) u.e[8 + i] = (__bf16)p[16 + kb + i];
  return u.v;
}

// B (32x16 bf16): lane holds col n=l&15; elems 0..7 -> K=kb..kb+7,
// elems 8..15 -> K=16+kb.., kb=(l>>4)*8.
// Source row-major [n][k].
__device__ __forceinline__ v16bf load_b_bf16(const __bf16* __restrict__ base,
                                             int col0, int ld, int k0, int lane) {
  int n  = lane & 15;
  int kb = (lane >> 4) * 8;
  const __bf16* p = base + (size_t)(col0 + n) * ld + k0 + kb;
  union { v16bf v; v8bf h[2]; } u;
  u.h[0] = *(const v8bf*)(p);
  u.h[1] = *(const v8bf*)(p + 16);
  return u.v;
}

// A fragment from LDS-staged P tile [16][256] bf16, row-major.
__device__ __forceinline__ v16bf load_a_lds(const __bf16* base, int k0, int lane) {
  int m  = lane & 15;
  int kb = (lane >> 4) * 8;
  const __bf16* p = base + m * 256 + k0;
  union { v16bf v; v8bf h[2]; } u;
  u.h[0] = *(const v8bf*)(p + kb);
  u.h[1] = *(const v8bf*)(p + 16 + kb);
  return u.v;
}

// ---------------------------------------------------------------------------
__global__ void cvt_f32_bf16_kernel(const float* __restrict__ in,
                                    __bf16* __restrict__ out, int n) {
  int i = blockIdx.x * blockDim.x + threadIdx.x;
  if (i < n) out[i] = (__bf16)in[i];
}

// Y = X @ W^T. layout 0: out[b,h,s,d] (Q,K). layout 1: out[b,h,d,s] (V^T).
// grid (64,16), block 256 (8 waves, wave tile 16x64).
__global__ void qkv_proj_kernel(const float* __restrict__ X,
                                const __bf16* __restrict__ W,
                                __bf16* __restrict__ out, int layout) {
  const int K = 1024, S = 2048, H = 16;
  int lane = threadIdx.x & 31;
  int wave = threadIdx.x >> 5;
  int row0 = blockIdx.x * 128 + wave * 16;
  int h    = blockIdx.y;
  int col0 = h * 64;

  v8f acc[4] = {};
  for (int k0 = 0; k0 < K; k0 += 32) {
    if (k0 + 32 < K)  // prefetch next weight chunk (global_prefetch_b8)
      __builtin_prefetch(W + (size_t)(col0 + (lane & 15)) * K + k0 + 32, 0, 1);
    v16bf a = load_a_f32(X, row0, K, k0, lane);
#pragma unroll
    for (int j = 0; j < 4; ++j) {
      v16bf b = load_b_bf16(W, col0 + 16 * j, K, k0, lane);
      acc[j] = wmma_bf16(a, b, acc[j]);
    }
  }
#pragma unroll
  for (int j = 0; j < 4; ++j) {
    int d = 16 * j + (lane & 15);
#pragma unroll
    for (int i = 0; i < 8; ++i) {
      int m = row0 + (lane >> 4) * 8 + i;
      int b = m >> 11;
      int s = m & 2047;
      size_t idx;
      if (layout == 0) idx = ((size_t)(b * H + h) * S + s) * 64 + d;
      else             idx = ((size_t)(b * H + h) * 64 + d) * (size_t)S + s;
      out[idx] = (__bf16)acc[j][i];
    }
  }
}

// Fused scores + softmax + context.
// grid (S/16, B*H) = (128, 64); block = 256 (8 waves).
// Block handles 16 query rows of one (b,h); wave w owns keys [256w, 256w+256).
// Dynamic LDS: 66KB = 8 waves * 8KB (P tile bf16 [16][256], then reused as the
// f32 [16][64] context partial) + 2KB Q tile staged by the Tensor Data Mover.
// Softmax-reduction scratch aliases wave 0's P region (used before P staging,
// protected by __syncthreads).
__global__ void fused_attn_kernel(const __bf16* __restrict__ Qb,
                                  const __bf16* __restrict__ Kb,
                                  const __bf16* __restrict__ Vt,
                                  float* __restrict__ attn,
                                  __bf16* __restrict__ ctx) {
  const int S = 2048, Dh = 64;
  extern __shared__ char smem[];

  int lane = threadIdx.x & 31;
  int w    = threadIdx.x >> 5;
  int bh   = blockIdx.y;
  int b    = bh >> 4, h = bh & 15;
  int q0   = blockIdx.x * 16;
  int ks0  = w * 256;

  const __bf16* Qh = Qb + (size_t)bh * S * Dh;
  const __bf16* Kh = Kb + (size_t)bh * S * Dh;
  const __bf16* Vh = Vt + (size_t)bh * Dh * S;
  float*        ah = attn + (size_t)bh * S * S;

  float* red_max = (float*)smem;           // [8][16]
  float* red_sum = (float*)(smem + 512);   // [8][16]

  // ---- phase 0: TDM-stage the 16x64 bf16 Q tile into LDS (one wave) ----
  // D# per CDNA5 ISA 8.3/8.4: 2-D tensor, data_size=2B, tile 64x16,
  // row stride 64 elements, dest LDS byte offset 65536.
  if (w == 0) {
    unsigned long long ga =
        (unsigned long long)(const void*)(Qh + (size_t)q0 * Dh);
    v4u g0;
    g0[0] = 1u;                                        // count=1, user mode
    g0[1] = 65536u;                                    // lds_addr (bytes)
    g0[2] = (unsigned)(ga & 0xFFFFFFFFu);              // global_addr[31:0]
    g0[3] = (unsigned)((ga >> 32) & 0x01FFFFFFu)       // global_addr[56:32]
            | 0x80000000u;                             // type=2 ("image")
    v8i g1;
    g1[0] = 0x10000;        // workgroup_mask=0, data_size=1 (2 bytes)
    g1[1] = 64 << 16;       // tensor_dim0 = 64 (bits 79:48 low half)
    g1[2] = 2048 << 16;     // tensor_dim0 hi=0 | tensor_dim1 = 2048 (low half)
    g1[3] = 64 << 16;       // tensor_dim1 hi=0 | tile_dim0 = 64
    g1[4] = 16;             // tile_dim1 = 16, tile_dim2 = 0
    g1[5] = 64;             // tensor_dim0_stride[31:0] = 64 elements
    g1[6] = 0;              // stride hi / tensor_dim1_stride lo
    g1[7] = 0;
    v4i gz4 = {0, 0, 0, 0};           // groups 2/3 unused (2-D tensor)
    v8i gz8 = {0, 0, 0, 0, 0, 0, 0, 0};
    __builtin_amdgcn_tensor_load_to_lds(g0, g1, gz4, gz4, gz8, 0);
    __builtin_amdgcn_s_wait_tensorcnt(0);
  }
  __syncthreads();
  const __bf16* Qlds = (const __bf16*)(smem + 65536);  // [16][64]

  // ---- phase 1: scores for this wave's 256-key slice (32 WMMAs) ----
  v8f sc[16] = {};
  v16bf aq0 = load_a_bf16(Qlds, 0, Dh, 0, lane);
  v16bf aq1 = load_a_bf16(Qlds, 0, Dh, 32, lane);
#pragma unroll
  for (int j = 0; j < 16; ++j) {
    v16bf b0 = load_b_bf16(Kh, ks0 + 16 * j, Dh, 0, lane);
    sc[j] = wmma_bf16(aq0, b0, sc[j]);
    v16bf b1 = load_b_bf16(Kh, ks0 + 16 * j, Dh, 32, lane);
    sc[j] = wmma_bf16(aq1, b1, sc[j]);
  }
#pragma unroll
  for (int j = 0; j < 16; ++j)
#pragma unroll
    for (int i = 0; i < 8; ++i) sc[j][i] *= 0.125f;   // 1/sqrt(64)

  // ---- phase 2: row max (lane holds rows (l>>4)*8+i, col residue l&15) ----
  float rmax[8];
#pragma unroll
  for (int i = 0; i < 8; ++i) {
    float m = sc[0][i];
#pragma unroll
    for (int j = 1; j < 16; ++j) m = fmaxf(m, sc[j][i]);
#pragma unroll
    for (int mk = 1; mk < 16; mk <<= 1) m = fmaxf(m, __shfl_xor(m, mk, 32));
    rmax[i] = m;
  }
  if ((lane & 15) == 0) {
    int rb = (lane >> 4) * 8;
#pragma unroll
    for (int i = 0; i < 8; ++i) red_max[w * 16 + rb + i] = rmax[i];
  }
  __syncthreads();
  float gmax[8];
#pragma unroll
  for (int i = 0; i < 8; ++i) {
    int r = (lane >> 4) * 8 + i;
    float m = red_max[r];
#pragma unroll
    for (int w2 = 1; w2 < 8; ++w2) m = fmaxf(m, red_max[w2 * 16 + r]);
    gmax[i] = m;
  }

  // ---- phase 3: exp and row sum ----
  float rsum[8] = {0, 0, 0, 0, 0, 0, 0, 0};
#pragma unroll
  for (int j = 0; j < 16; ++j)
#pragma unroll
    for (int i = 0; i < 8; ++i) {
      float e = __expf(sc[j][i] - gmax[i]);
      sc[j][i] = e;
      rsum[i] += e;
    }
#pragma unroll
  for (int i = 0; i < 8; ++i)
#pragma unroll
    for (int mk = 1; mk < 16; mk <<= 1) rsum[i] += __shfl_xor(rsum[i], mk, 32);
  if ((lane & 15) == 0) {
    int rb = (lane >> 4) * 8;
#pragma unroll
    for (int i = 0; i < 8; ++i) red_sum[w * 16 + rb + i] = rsum[i];
  }
  __syncthreads();
  float inv[8];
#pragma unroll
  for (int i = 0; i < 8; ++i) {
    int r = (lane >> 4) * 8 + i;
    float s = red_sum[r];
#pragma unroll
    for (int w2 = 1; w2 < 8; ++w2) s += red_sum[w2 * 16 + r];
    inv[i] = 1.0f / s;
  }
  __syncthreads();   // everyone done with red_* before P staging clobbers it

  // ---- phase 4: normalized P -> global attn (once) + LDS (bf16) ----
  __bf16* Pw = (__bf16*)(smem + (size_t)w * 8192);   // [16][256]
#pragma unroll
  for (int j = 0; j < 16; ++j) {
    int c = 16 * j + (lane & 15);
#pragma unroll
    for (int i = 0; i < 8; ++i) {
      int r = (lane >> 4) * 8 + i;
      float p = sc[j][i] * inv[i];
      Pw[r * 256 + c] = (__bf16)p;
      ah[(size_t)(q0 + r) * S + ks0 + c] = p;
    }
  }

  // ---- phase 5: context partial: P(16x256) @ V(256x64) (32 WMMAs) ----
  v8f cacc[4] = {};
#pragma unroll
  for (int k0 = 0; k0 < 256; k0 += 32) {
    v16bf a = load_a_lds(Pw, k0, lane);
#pragma unroll
    for (int j = 0; j < 4; ++j) {
      v16bf bf = load_b_bf16(Vh, 16 * j, S, ks0 + k0, lane);
      cacc[j] = wmma_bf16(a, bf, cacc[j]);
    }
  }

  // ---- phase 6: reduce partials over 8 waves; ctx -> global bf16 ----
  float* Cw = (float*)(smem + (size_t)w * 8192);     // [16][64], reuses P region
#pragma unroll
  for (int j = 0; j < 4; ++j)
#pragma unroll
    for (int i = 0; i < 8; ++i)
      Cw[((lane >> 4) * 8 + i) * 64 + 16 * j + (lane & 15)] = cacc[j][i];
  __syncthreads();
#pragma unroll
  for (int e = 0; e < 4; ++e) {
    int idx = threadIdx.x + 256 * e;   // 0..1023 -> (r,d)
    int r = idx >> 6, d = idx & 63;
    float s = 0.f;
#pragma unroll
    for (int w2 = 0; w2 < 8; ++w2)
      s += ((const float*)(smem + (size_t)w2 * 8192))[r * 64 + d];
    ctx[((size_t)(b * S) + q0 + r) * 1024 + h * 64 + d] = (__bf16)s;
  }
}

// pre = ctx @ Wfc^T. grid (64,16).
__global__ void outproj_kernel(const __bf16* __restrict__ A,
                               const __bf16* __restrict__ W,
                               float* __restrict__ out) {
  const int K = 1024;
  int lane = threadIdx.x & 31;
  int wave = threadIdx.x >> 5;
  int row0 = blockIdx.x * 128 + wave * 16;
  int col0 = blockIdx.y * 64;

  v8f acc[4] = {};
  for (int k0 = 0; k0 < K; k0 += 32) {
    if (k0 + 32 < K)
      __builtin_prefetch(W + (size_t)(col0 + (lane & 15)) * K + k0 + 32, 0, 1);
    v16bf a = load_a_bf16(A, row0, K, k0, lane);
#pragma unroll
    for (int j = 0; j < 4; ++j) {
      v16bf b = load_b_bf16(W, col0 + 16 * j, K, k0, lane);
      acc[j] = wmma_bf16(a, b, acc[j]);
    }
  }
#pragma unroll
  for (int j = 0; j < 4; ++j) {
    int n = col0 + 16 * j + (lane & 15);
#pragma unroll
    for (int i = 0; i < 8; ++i) {
      int m = row0 + (lane >> 4) * 8 + i;
      out[(size_t)m * 1024 + n] = acc[j][i];
    }
  }
}

// out = LayerNorm(pre + resid), D=1024. 1 block (256 thr) per row.
__global__ void ln_kernel(const float* __restrict__ pre,
                          const float* __restrict__ resid,
                          float* __restrict__ out) {
  const int D = 1024;
  __shared__ float sm[256];
  size_t row = blockIdx.x;
  const float* a = pre   + row * D;
  const float* r = resid + row * D;
  float*       o = out   + row * D;
  int t = threadIdx.x;
  float v[4];
  float s = 0.f;
#pragma unroll
  for (int i = 0; i < 4; ++i) { v[i] = a[t + 256 * i] + r[t + 256 * i]; s += v[i]; }
  sm[t] = s; __syncthreads();
  for (int off = 128; off > 0; off >>= 1) {
    if (t < off) sm[t] += sm[t + off];
    __syncthreads();
  }
  float mean = sm[0] * (1.0f / D); __syncthreads();
  float q = 0.f;
#pragma unroll
  for (int i = 0; i < 4; ++i) { float d = v[i] - mean; q += d * d; }
  sm[t] = q; __syncthreads();
  for (int off = 128; off > 0; off >>= 1) {
    if (t < off) sm[t] += sm[t + off];
    __syncthreads();
  }
  float rstd = rsqrtf(sm[0] * (1.0f / D) + 1e-5f);
#pragma unroll
  for (int i = 0; i < 4; ++i) o[t + 256 * i] = (v[i] - mean) * rstd;
}

// ---------------------------------------------------------------------------
extern "C" void kernel_launch(void* const* d_in, const int* in_sizes, int n_in,
                              void* d_out, int out_size, void* d_ws, size_t ws_size,
                              hipStream_t stream) {
  const float* x_q  = (const float*)d_in[0];
  const float* x_k  = (const float*)d_in[1];
  const float* x_v  = (const float*)d_in[2];
  const float* W_Q  = (const float*)d_in[3];
  const float* W_K  = (const float*)d_in[4];
  const float* W_V  = (const float*)d_in[5];
  const float* W_fc = (const float*)d_in[6];

  const int B = 4, S = 2048, D = 1024, H = 16;
  const size_t M  = (size_t)B * S;   // 8192
  const size_t MD = M * D;           // 8,388,608

  char* ws = (char*)d_ws;
  __bf16* Wqb  = (__bf16*)ws;  ws += (size_t)D * D * 2;
  __bf16* Wkb  = (__bf16*)ws;  ws += (size_t)D * D * 2;
  __bf16* Wvb  = (__bf16*)ws;  ws += (size_t)D * D * 2;
  __bf16* Wfcb = (__bf16*)ws;  ws += (size_t)D * D * 2;
  __bf16* Qb   = (__bf16*)ws;  ws += MD * 2;   // [b,h,s,d]
  __bf16* Kb   = (__bf16*)ws;  ws += MD * 2;   // [b,h,s,d]
  __bf16* Vt   = (__bf16*)ws;  ws += MD * 2;   // [b,h,d,s]
  __bf16* ctxb = (__bf16*)ws;  ws += MD * 2;   // [b*s, h*d]
  float*  pre  = (float*)ws;   ws += MD * 4;

  float* out_final = (float*)d_out;
  float* attn      = (float*)d_out + MD;

  dim3 blk(256);
  int nW = D * D;
  cvt_f32_bf16_kernel<<<nW / 256, blk, 0, stream>>>(W_Q,  Wqb,  nW);
  cvt_f32_bf16_kernel<<<nW / 256, blk, 0, stream>>>(W_K,  Wkb,  nW);
  cvt_f32_bf16_kernel<<<nW / 256, blk, 0, stream>>>(W_V,  Wvb,  nW);
  cvt_f32_bf16_kernel<<<nW / 256, blk, 0, stream>>>(W_fc, Wfcb, nW);

  qkv_proj_kernel<<<dim3(64, 16), blk, 0, stream>>>(x_q, Wqb, Qb, 0);
  qkv_proj_kernel<<<dim3(64, 16), blk, 0, stream>>>(x_k, Wkb, Kb, 0);
  qkv_proj_kernel<<<dim3(64, 16), blk, 0, stream>>>(x_v, Wvb, Vt, 1);

  // fused scores+softmax+context: 66KB dynamic LDS per block
  // (64KB P/ctx staging + 2KB TDM-loaded Q tile)
  fused_attn_kernel<<<dim3(S / 16, B * H), blk, 67584, stream>>>(Qb, Kb, Vt,
                                                                 attn, ctxb);

  outproj_kernel<<<dim3(64, 16), blk, 0, stream>>>(ctxb, Wfcb, pre);
  ln_kernel<<<M, blk, 0, stream>>>(pre, x_q, out_final);
}
